// Net_10393820857080
// MI455X (gfx1250) — matrix operand order repaired and checked by reading, not compiled, not run on hardware
//
#include <hip/hip_runtime.h>

#define GG 8
#define NN 1024
#define EE 16384
#define KP1 512
#define KP2 256

typedef __attribute__((ext_vector_type(16))) __bf16 v16bf;
typedef __attribute__((ext_vector_type(8)))  float  v8f;

union FragB16 { uint4 u[2]; v16bf v; };

// ---------------- utility kernels ----------------

__global__ void k_f32_to_bf16(const float* __restrict__ in, __bf16* __restrict__ out, int n) {
  int t = blockIdx.x * blockDim.x + threadIdx.x;
  if (t < n) out[t] = (__bf16)in[t];
}

__global__ void k_zero_f32(float* p, int n) {
  int t = blockIdx.x * blockDim.x + threadIdx.x;
  if (t < n) p[t] = 0.0f;
}

// Build Wcat = [B0|B1|B2|B3|B_bias|root] (cin x 384) directly in WMMA B-fragment
// swizzled order: chunk(kb, tile) = 32 lanes x 8 packed dwords, so each lane's
// fragment is two contiguous uint4 loads.
// Logical element: B[k][col], k = kb*32 + 2*j + hi*16 (+1 for high half of dword),
// col = tile*16 + lo.
__global__ void k_build_wcat_sw(const float* __restrict__ w2, const float* __restrict__ b2,
                                const float* __restrict__ root, __bf16* __restrict__ bsw,
                                int cin) {
  const int ntn = 24;                       // 384/16 column tiles
  int dd = blockIdx.x * blockDim.x + threadIdx.x;   // packed-dword index
  int total = (cin / 32) * ntn * 32 * 8;
  if (dd >= total) return;
  int j    = dd & 7;
  int lane = (dd >> 3) & 31;
  int tile = (dd >> 8) % ntn;
  int kb   = (dd >> 8) / ntn;
  int lo = lane & 15, hi = lane >> 4;
  int k   = kb * 32 + 2 * j + hi * 16;
  int col = tile * 16 + lo;
  auto src = [&](int i) -> float {
    if (col < 256) { int kk = col >> 6, o = col & 63; return w2[kk * (cin * 64) + i * 64 + o]; }
    if (col < 320) return b2[i * 64 + (col - 256)];
    return root[i * 64 + (col - 320)];
  };
  bsw[2 * dd]     = (__bf16)src(k);
  bsw[2 * dd + 1] = (__bf16)src(k + 1);
}

// h[e][k] = relu(attr_e * w1[k] + b1[k]), k = 0..3
__global__ void k_edge_hidden(const float* __restrict__ eattr, const float* __restrict__ w1,
                              const float* __restrict__ b1, float* __restrict__ h, int E) {
  int e = blockIdx.x * blockDim.x + threadIdx.x;
  if (e >= E) return;
  float a = eattr[e];
#pragma unroll
  for (int k = 0; k < 4; ++k) {
    float v = a * w1[k] + b1[k];
    h[e * 4 + k] = v > 0.f ? v : 0.f;
  }
}

// ---------------- WMMA GEMM: C[M,384] = A[M,K](bf16, row-major) @ B(swizzled bf16) ----------
// One wave computes a 16x64 strip: 1 A-fragment reused across 4 column tiles.
// K templated so the k-loop fully unrolls (K/32 * 4 wmma per wave).
template <int K>
__global__ void k_gemm_bf16(const __bf16* __restrict__ A, const __bf16* __restrict__ Bsw,
                            float* __restrict__ C, int M) {
  const int Nt = 384;
  const int ntn = 24;        // 16-col tiles
  const int ntn4 = 6;        // groups of 4 tiles
  int gw = (blockIdx.x * blockDim.x + threadIdx.x) >> 5;
  int tm = gw / ntn4, tc = gw % ntn4;
  if (tm * 16 >= M) return;
  int lane = threadIdx.x & 31;
  int lo = lane & 15, hi = lane >> 4;
  v8f acc[4] = {};
  const __bf16* arow = A + (size_t)(tm * 16 + lo) * K;
#pragma unroll
  for (int kb = 0; kb < K / 32; ++kb) {
    FragB16 a;
    a.u[0] = *(const uint4*)(arow + kb * 32 + hi * 16);
    a.u[1] = *(const uint4*)(arow + kb * 32 + hi * 16 + 8);
#pragma unroll
    for (int t = 0; t < 4; ++t) {
      int tile = tc * 4 + t;
      const uint4* bp =
          (const uint4*)(Bsw + (((size_t)(kb * ntn + tile) * 32 + lane) << 4));
      FragB16 b;
      b.u[0] = bp[0];
      b.u[1] = bp[1];
      acc[t] = __builtin_amdgcn_wmma_f32_16x16x32_bf16(false, a.v, false, b.v, (short)0,
                                                       acc[t], false, false);
    }
  }
#pragma unroll
  for (int t = 0; t < 4; ++t) {
    float* crow = C + (size_t)(tm * 16 + hi * 8) * Nt + (tc * 4 + t) * 16 + lo;
#pragma unroll
    for (int r = 0; r < 8; ++r) crow[(size_t)r * Nt] = acc[t][r];
  }
}

// ---------------- NNConv scatter: msg_e = (Σk h_ek * Y[src, k*64+o]) + Y[src,256+o] --------
__global__ void k_nnconv_scatter(const float* __restrict__ Y, const float* __restrict__ hE,
                                 const int* __restrict__ src, const int* __restrict__ dst,
                                 const float* __restrict__ valid,
                                 float* __restrict__ sums, float* __restrict__ cnt, int E) {
  int t = blockIdx.x * blockDim.x + threadIdx.x;
  int e = t >> 6, o = t & 63;
  if (e >= E) return;
  float v = valid ? valid[e] : 1.0f;
  int s = src[e], d = dst[e];
  const float* y = Y + (size_t)s * 384;
  float m = y[256 + o];
#pragma unroll
  for (int k = 0; k < 4; ++k) m += hE[e * 4 + k] * y[k * 64 + o];
  m *= v;
  atomicAdd(&sums[(size_t)d * 64 + o], m);
  if (o == 0) atomicAdd(&cnt[d], v);
}

// out = relu(sums/max(cnt,1) + Y[:,320:384] + bias)
__global__ void k_conv_finalize(const float* __restrict__ sums, const float* __restrict__ cnt,
                                const float* __restrict__ Y, const float* __restrict__ bias,
                                float* __restrict__ out, int n_nodes) {
  int t = blockIdx.x * blockDim.x + threadIdx.x;
  int node = t >> 6, o = t & 63;
  if (node >= n_nodes) return;
  float c = cnt[node];
  c = c > 1.f ? c : 1.f;
  float v = sums[(size_t)node * 64 + o] / c + Y[(size_t)node * 384 + 320 + o] + bias[o];
  out[(size_t)node * 64 + o] = v > 0.f ? v : 0.f;
}

// ---------------- TopK pooling: per-graph bitonic sort of tanh(h·w/||w||) ----------------
__global__ void k_topk_pool(const float* __restrict__ h, const float* __restrict__ w,
                            int n_per, int k, float* __restrict__ x_new,
                            __bf16* __restrict__ x_new_b, int* __restrict__ mapping) {
  __shared__ float key[1024];
  __shared__ int   sidx[1024];
  int g = blockIdx.x;
  int tid = threadIdx.x;
  float nw = 0.f;
  for (int i = 0; i < 64; ++i) { float t = w[i]; nw += t * t; }
  nw = sqrtf(nw);
  for (int i = tid; i < n_per; i += blockDim.x) {
    const float* row = h + (size_t)(g * n_per + i) * 64;
    float dv = 0.f;
    for (int o = 0; o < 64; ++o) dv += row[o] * w[o];
    key[i]  = tanhf(dv / nw);
    sidx[i] = i;
    mapping[g * n_per + i] = -1;
  }
  __syncthreads();
  for (int kk = 2; kk <= n_per; kk <<= 1) {
    for (int j = kk >> 1; j > 0; j >>= 1) {
      for (int t = tid; t < n_per; t += blockDim.x) {
        int ixj = t ^ j;
        if (ixj > t) {
          bool desc = ((t & kk) == 0);   // descending overall
          float a = key[t], b = key[ixj];
          if (desc ? (a < b) : (a > b)) {
            key[t] = b; key[ixj] = a;
            int ti = sidx[t]; sidx[t] = sidx[ixj]; sidx[ixj] = ti;
          }
        }
      }
      __syncthreads();
    }
  }
  for (int jj = tid; jj < k; jj += blockDim.x) {
    int   old = sidx[jj];
    float val = key[jj];
    int nr = g * k + jj;
    mapping[g * n_per + old] = nr;
    const float* row = h + (size_t)(g * n_per + old) * 64;
    for (int o = 0; o < 64; ++o) {
      float vv = row[o] * val;
      x_new[(size_t)nr * 64 + o]   = vv;
      x_new_b[(size_t)nr * 64 + o] = (__bf16)vv;
    }
  }
}

__global__ void k_relabel(const int* __restrict__ src, const int* __restrict__ dst,
                          const float* __restrict__ vprev, const int* __restrict__ mapping,
                          int* __restrict__ s_out, int* __restrict__ d_out,
                          float* __restrict__ v_out, int E) {
  int e = blockIdx.x * blockDim.x + threadIdx.x;
  if (e >= E) return;
  int ns = mapping[src[e]], nd = mapping[dst[e]];
  bool keep = (ns >= 0) && (nd >= 0);
  float v = (vprev ? vprev[e] : 1.f) * (keep ? 1.f : 0.f);
  s_out[e] = keep ? ns : 0;
  d_out[e] = keep ? nd : 0;
  v_out[e] = v;
}

// out[g][0:64] = max over k nodes, out[g][64:128] = mean
__global__ void k_pool_cat(const float* __restrict__ hp, int k, float* __restrict__ out) {
  int g = blockIdx.x, o = threadIdx.x;   // 64 threads
  float mx = -INFINITY, sm = 0.f;
  for (int j = 0; j < k; ++j) {
    float v = hp[(size_t)(g * k + j) * 64 + o];
    mx = fmaxf(mx, v);
    sm += v;
  }
  out[g * 128 + o]      = mx;
  out[g * 128 + 64 + o] = sm / (float)k;
}

// ---------------- MLP head: [8,256] -> 64 -> 8 -> 2 -> log_softmax ----------------
__global__ void k_head(const float* __restrict__ x1, const float* __restrict__ x2,
                       const float* __restrict__ fc1w, const float* __restrict__ fc1b,
                       const float* __restrict__ fc2w, const float* __restrict__ fc2b,
                       const float* __restrict__ fc3w, const float* __restrict__ fc3b,
                       float* __restrict__ out) {
  __shared__ float z[8][256];
  __shared__ float z1[8][64];
  __shared__ float z2[8][8];
  __shared__ float z3[8][2];
  int tid = threadIdx.x;   // 64
  for (int t = tid; t < 8 * 128; t += 64) {
    int g = t / 128, c = t % 128;
    z[g][c]       = x1[g * 128 + c];
    z[g][128 + c] = x2[g * 128 + c];
  }
  __syncthreads();
  for (int t = tid; t < 8 * 64; t += 64) {
    int g = t / 64, j = t % 64;
    float s = fc1b[j];
    for (int i = 0; i < 256; ++i) s += z[g][i] * fc1w[i * 64 + j];
    z1[g][j] = s > 0.f ? s : 0.f;
  }
  __syncthreads();
  for (int t = tid; t < 8 * 8; t += 64) {
    int g = t / 8, j = t % 8;
    float s = fc2b[j];
    for (int i = 0; i < 64; ++i) s += z1[g][i] * fc2w[i * 8 + j];
    z2[g][j] = s > 0.f ? s : 0.f;
  }
  __syncthreads();
  for (int t = tid; t < 16; t += 64) {
    int g = t / 2, j = t % 2;
    float s = fc3b[j];
    for (int i = 0; i < 8; ++i) s += z2[g][i] * fc3w[i * 2 + j];
    z3[g][j] = s;
  }
  __syncthreads();
  for (int t = tid; t < 16; t += 64) {
    int g = t / 2, j = t % 2;
    float a = z3[g][0], b = z3[g][1];
    float m = fmaxf(a, b);
    float lse = m + logf(expf(a - m) + expf(b - m));
    out[g * 2 + j] = z3[g][j] - lse;
  }
}

// ---------------- launch ----------------

extern "C" void kernel_launch(void* const* d_in, const int* in_sizes, int n_in,
                              void* d_out, int out_size, void* d_ws, size_t ws_size,
                              hipStream_t stream) {
  (void)in_sizes; (void)n_in; (void)out_size; (void)ws_size;
  const float* x          = (const float*)d_in[0];
  const int*   esrc       = (const int*)d_in[1];
  const int*   edst       = (const int*)d_in[2];
  const float* eattr      = (const float*)d_in[3];
  const float* n1_w1      = (const float*)d_in[4];
  const float* n1_b1      = (const float*)d_in[5];
  const float* n1_w2      = (const float*)d_in[6];
  const float* n1_b2      = (const float*)d_in[7];
  const float* conv1_root = (const float*)d_in[8];
  const float* conv1_bias = (const float*)d_in[9];
  const float* pool1_w    = (const float*)d_in[10];
  const float* n2_w1      = (const float*)d_in[11];
  const float* n2_b1      = (const float*)d_in[12];
  const float* n2_w2      = (const float*)d_in[13];
  const float* n2_b2      = (const float*)d_in[14];
  const float* conv2_root = (const float*)d_in[15];
  const float* conv2_bias = (const float*)d_in[16];
  const float* pool2_w    = (const float*)d_in[17];
  const float* fc1_w      = (const float*)d_in[18];
  const float* fc1_b      = (const float*)d_in[19];
  const float* fc2_w      = (const float*)d_in[20];
  const float* fc2_b      = (const float*)d_in[21];
  const float* fc3_w      = (const float*)d_in[22];
  const float* fc3_b      = (const float*)d_in[23];

  char* wsp = (char*)d_ws;
  size_t off = 0;
  auto alloc = [&](size_t b) -> void* {
    void* p = wsp + off;
    off = (off + b + 255) & ~(size_t)255;
    return p;
  };
  __bf16* xb       = (__bf16*)alloc((size_t)GG * NN * 128 * 2);
  __bf16* wsw1     = (__bf16*)alloc((size_t)128 * 384 * 2);
  __bf16* wsw2     = (__bf16*)alloc((size_t)64 * 384 * 2);
  float*  h1e      = (float*)alloc((size_t)EE * 4 * 4);
  float*  h2e      = (float*)alloc((size_t)EE * 4 * 4);
  float*  Y1       = (float*)alloc((size_t)GG * NN * 384 * 4);
  float*  sums1    = (float*)alloc((size_t)GG * NN * 64 * 4);
  float*  cnt1     = (float*)alloc((size_t)GG * NN * 4);
  float*  conv1out = (float*)alloc((size_t)GG * NN * 64 * 4);
  int*    map1     = (int*)alloc((size_t)GG * NN * 4);
  int*    s1       = (int*)alloc((size_t)EE * 4);
  int*    d1       = (int*)alloc((size_t)EE * 4);
  float*  v1       = (float*)alloc((size_t)EE * 4);
  float*  xp1      = (float*)alloc((size_t)GG * KP1 * 64 * 4);
  __bf16* xp1b     = (__bf16*)alloc((size_t)GG * KP1 * 64 * 2);
  float*  x1cat    = (float*)alloc((size_t)GG * 128 * 4);
  float*  Y2       = (float*)alloc((size_t)GG * KP1 * 384 * 4);
  float*  sums2    = (float*)alloc((size_t)GG * KP1 * 64 * 4);
  float*  cnt2     = (float*)alloc((size_t)GG * KP1 * 4);
  float*  conv2out = (float*)alloc((size_t)GG * KP1 * 64 * 4);
  int*    map2     = (int*)alloc((size_t)GG * KP1 * 4);
  float*  xp2      = (float*)alloc((size_t)GG * KP2 * 64 * 4);
  __bf16* xp2b     = (__bf16*)alloc((size_t)GG * KP2 * 64 * 2);
  float*  x2cat    = (float*)alloc((size_t)GG * 128 * 4);

  const int nN1 = GG * NN;        // 8192
  const int nN2 = GG * KP1;       // 4096

  // prep
  k_f32_to_bf16<<<(nN1 * 128 + 255) / 256, 256, 0, stream>>>(x, xb, nN1 * 128);
  k_build_wcat_sw<<<((128 / 32) * 24 * 256 + 255) / 256, 256, 0, stream>>>(
      n1_w2, n1_b2, conv1_root, wsw1, 128);
  k_build_wcat_sw<<<((64 / 32) * 24 * 256 + 255) / 256, 256, 0, stream>>>(
      n2_w2, n2_b2, conv2_root, wsw2, 64);
  k_edge_hidden<<<(EE + 255) / 256, 256, 0, stream>>>(eattr, n1_w1, n1_b1, h1e, EE);
  k_edge_hidden<<<(EE + 255) / 256, 256, 0, stream>>>(eattr, n2_w1, n2_b1, h2e, EE);

  // conv1: Y1 = xb[8192,128] @ Wcat1[128,384]   (WMMA, 16 v_wmma per wave)
  {
    int waves = (nN1 / 16) * 6;          // 3072 waves, 8 per block
    k_gemm_bf16<128><<<waves / 8, 256, 0, stream>>>(xb, wsw1, Y1, nN1);
  }
  k_zero_f32<<<(nN1 * 64 + 255) / 256, 256, 0, stream>>>(sums1, nN1 * 64);
  k_zero_f32<<<(nN1 + 255) / 256, 256, 0, stream>>>(cnt1, nN1);
  k_nnconv_scatter<<<(EE * 64 + 255) / 256, 256, 0, stream>>>(Y1, h1e, esrc, edst,
                                                              (const float*)nullptr,
                                                              sums1, cnt1, EE);
  k_conv_finalize<<<(nN1 * 64 + 255) / 256, 256, 0, stream>>>(sums1, cnt1, Y1, conv1_bias,
                                                              conv1out, nN1);
  // pool1
  k_topk_pool<<<GG, 512, 0, stream>>>(conv1out, pool1_w, NN, KP1, xp1, xp1b, map1);
  k_relabel<<<(EE + 255) / 256, 256, 0, stream>>>(esrc, edst, (const float*)nullptr,
                                                  map1, s1, d1, v1, EE);
  k_pool_cat<<<GG, 64, 0, stream>>>(xp1, KP1, x1cat);

  // conv2: Y2 = xp1b[4096,64] @ Wcat2[64,384]   (WMMA)
  {
    int waves = (nN2 / 16) * 6;          // 1536 waves
    k_gemm_bf16<64><<<waves / 8, 256, 0, stream>>>(xp1b, wsw2, Y2, nN2);
  }
  k_zero_f32<<<(nN2 * 64 + 255) / 256, 256, 0, stream>>>(sums2, nN2 * 64);
  k_zero_f32<<<(nN2 + 255) / 256, 256, 0, stream>>>(cnt2, nN2);
  k_nnconv_scatter<<<(EE * 64 + 255) / 256, 256, 0, stream>>>(Y2, h2e, s1, d1, v1,
                                                              sums2, cnt2, EE);
  k_conv_finalize<<<(nN2 * 64 + 255) / 256, 256, 0, stream>>>(sums2, cnt2, Y2, conv2_bias,
                                                              conv2out, nN2);
  // pool2 (s2/d2/v2 are dead values in the reference; skip relabel)
  k_topk_pool<<<GG, 512, 0, stream>>>(conv2out, pool2_w, KP1, KP2, xp2, xp2b, map2);
  k_pool_cat<<<GG, 64, 0, stream>>>(xp2, KP2, x2cat);

  // head
  k_head<<<1, 64, 0, stream>>>(x1cat, x2cat, fc1_w, fc1_b, fc2_w, fc2_b, fc3_w, fc3_b,
                               (float*)d_out);
}
